// GraphMolMasker_9251359555632
// MI455X (gfx1250) — compile-verified, matrix operand order checked by reading, not run on hardware
//
#include <hip/hip_runtime.h>
#include <hip/hip_bf16.h>
#include <math.h>

#define NN 100000
#define EE 600000
#define GG 4096
#define DD 128
#define AV 100
#define BV 8

typedef __attribute__((ext_vector_type(16))) __bf16 v16bf;
typedef __attribute__((ext_vector_type(4)))  __bf16 v4bf;
typedef __attribute__((ext_vector_type(8)))  float  v8f;

__device__ __forceinline__ float sigmoidf_(float x) { return 1.0f / (1.0f + __expf(-x)); }

// ---------------------------------------------------------------- fill
__global__ void fill_kernel(float* __restrict__ p, int n, float v) {
    int i = blockIdx.x * blockDim.x + threadIdx.x;
    if (i < n) p[i] = v;
}

// ---------------------------------------------------------------- atom encoder: h[n,:] = sum_j atom_tables[j, x[n,j], :]
__global__ __launch_bounds__(256) void atom_encoder_kernel(
    const int* __restrict__ x, const float* __restrict__ atom_tables,
    float* __restrict__ h)
{
    int n = blockIdx.x * 8 + (threadIdx.x >> 5);
    if (n >= NN) return;
    int lane = threadIdx.x & 31;
    int c = lane * 4;
    float4 s = make_float4(0.f, 0.f, 0.f, 0.f);
#pragma unroll
    for (int j = 0; j < 9; ++j) {
        int idx = x[n * 9 + j];
        const float4 tv = *(const float4*)(atom_tables + ((size_t)(j * AV + idx)) * DD + c);
        s.x += tv.x; s.y += tv.y; s.z += tv.z; s.w += tv.w;
    }
    *(float4*)(h + (size_t)n * DD + c) = s;
}

// ---------------------------------------------------------------- fused bond-encoder + message + scatter-add
__global__ __launch_bounds__(256) void message_kernel(
    const float* __restrict__ h, const int* __restrict__ edge_index,
    const int* __restrict__ edge_attr, const float* __restrict__ bond_l,
    float* __restrict__ agg)
{
    int e = blockIdx.x * 8 + (threadIdx.x >> 5);
    if (e >= EE) return;
    int lane = threadIdx.x & 31;
    int c = lane * 4;
    int row = edge_index[e];
    int col = edge_index[EE + e];
    int a0 = edge_attr[e * 3 + 0];
    int a1 = edge_attr[e * 3 + 1];
    int a2 = edge_attr[e * 3 + 2];
    const float4 t0 = *(const float4*)(bond_l + (0 * BV + a0) * DD + c);
    const float4 t1 = *(const float4*)(bond_l + (1 * BV + a1) * DD + c);
    const float4 t2 = *(const float4*)(bond_l + (2 * BV + a2) * DD + c);
    const float4 hv = *(const float4*)(h + (size_t)row * DD + c);
    float m0 = fmaxf(hv.x + t0.x + t1.x + t2.x, 0.f);
    float m1 = fmaxf(hv.y + t0.y + t1.y + t2.y, 0.f);
    float m2 = fmaxf(hv.z + t0.z + t1.z + t2.z, 0.f);
    float m3 = fmaxf(hv.w + t0.w + t1.w + t2.w, 0.f);
    float* dst = agg + (size_t)col * DD + c;
    atomicAdd(dst + 0, m0);
    atomicAdd(dst + 1, m1);
    atomicAdd(dst + 2, m2);
    atomicAdd(dst + 3, m3);
}

// ---------------------------------------------------------------- pack W (K x ncols, f32) into bf16 WMMA B-fragment order
// layout: packB[(ntile*nk + ks)*32 + lane] = v16bf fragment for that wave-lane
__global__ void pack_b_kernel(const float* __restrict__ W, int ncols, int nk,
                              v16bf* __restrict__ packB)
{
    int tid = blockIdx.x * blockDim.x + threadIdx.x;
    int total = (ncols / 16) * nk * 32;
    if (tid >= total) return;
    int lane  = tid & 31;
    int ks    = (tid >> 5) % nk;
    int ntile = (tid >> 5) / nk;
    int n = ntile * 16 + (lane & 15);
    int half = lane >> 4;
    v16bf b;
#pragma unroll
    for (int v = 0; v < 8; ++v) {
        int kp = ks * 32 + ((v & 4) << 2) + half * 8 + (v & 3) * 2;
        b[2 * v]     = (__bf16)W[(size_t)kp * ncols + n];
        b[2 * v + 1] = (__bf16)W[(size_t)(kp + 1) * ncols + n];
    }
    packB[tid] = b;
}

// build A fragment from an LDS row-major bf16 tile (row stride = K elements)
__device__ __forceinline__ v16bf load_a_frag(const __bf16* __restrict__ As, int K,
                                             int mrow, int half, int kk)
{
    const __bf16* arow = As + mrow * K + kk + half * 8;
    v4bf p0 = *(const v4bf*)(arow);
    v4bf p1 = *(const v4bf*)(arow + 4);
    v4bf p2 = *(const v4bf*)(arow + 16);
    v4bf p3 = *(const v4bf*)(arow + 20);
    v16bf a;
    a[0] = p0[0];  a[1] = p0[1];  a[2] = p0[2];  a[3] = p0[3];
    a[4] = p1[0];  a[5] = p1[1];  a[6] = p1[2];  a[7] = p1[3];
    a[8] = p2[0];  a[9] = p2[1];  a[10] = p2[2]; a[11] = p2[3];
    a[12] = p3[0]; a[13] = p3[1]; a[14] = p3[2]; a[15] = p3[3];
    return a;
}

// ---------------------------------------------------------------- GEMM1: t[m,0:256] = ((1+eps)h[m,:] + agg[m,:]) @ W1 + b1
// one block per 16-row tile; A staged in LDS (bf16); 8 waves x 2 col-tiles.
// BatchNorm column sum / sum-of-squares reduced in-register at the epilogue.
__global__ __launch_bounds__(256) void gemm1_kernel(
    const float* __restrict__ h, const float* __restrict__ agg,
    const float* __restrict__ epsp, const v16bf* __restrict__ packB,
    const float* __restrict__ bias, float* __restrict__ out,
    float* __restrict__ csum, float* __restrict__ csq)
{
    __shared__ __bf16 As[16 * DD];   // 4 KB
    const int tid = threadIdx.x;
    const int lane = tid & 31;
    const int wave = tid >> 5;
    const int mtile = blockIdx.x;
    const float epsv = 1.0f + epsp[0];

    // stage A: 16x128 z_in tile -> bf16 LDS, 8 elements per thread
    {
        int e = tid * 8;
        int m = e >> 7, k = e & 127;
        const float* hp = h   + (size_t)(mtile * 16 + m) * DD + k;
        const float* ap = agg + (size_t)(mtile * 16 + m) * DD + k;
        float4 h0 = *(const float4*)(hp);
        float4 h1 = *(const float4*)(hp + 4);
        float4 a0 = *(const float4*)(ap);
        float4 a1 = *(const float4*)(ap + 4);
        __bf16* dst = As + m * DD + k;
        dst[0] = (__bf16)(epsv * h0.x + a0.x);
        dst[1] = (__bf16)(epsv * h0.y + a0.y);
        dst[2] = (__bf16)(epsv * h0.z + a0.z);
        dst[3] = (__bf16)(epsv * h0.w + a0.w);
        dst[4] = (__bf16)(epsv * h1.x + a1.x);
        dst[5] = (__bf16)(epsv * h1.y + a1.y);
        dst[6] = (__bf16)(epsv * h1.z + a1.z);
        dst[7] = (__bf16)(epsv * h1.w + a1.w);
    }
    __syncthreads();

    const int ntile0 = wave;
    const int ntile1 = wave + 8;
    const int mrow = lane & 15;
    const int half = lane >> 4;
    v8f acc0 = {}, acc1 = {};
#pragma unroll
    for (int ks = 0; ks < 4; ++ks) {
        v16bf a = load_a_frag(As, DD, mrow, half, ks * 32);
        v16bf b0 = packB[(ntile0 * 4 + ks) * 32 + lane];
        v16bf b1 = packB[(ntile1 * 4 + ks) * 32 + lane];
        acc0 = __builtin_amdgcn_wmma_f32_16x16x32_bf16(false, a, false, b0, (short)0, acc0, false, false);
        acc1 = __builtin_amdgcn_wmma_f32_16x16x32_bf16(false, a, false, b1, (short)0, acc1, false, false);
    }
    const int n0 = ntile0 * 16 + (lane & 15);
    const int n1 = ntile1 * 16 + (lane & 15);
    const float bv0 = bias[n0];
    const float bv1 = bias[n1];
    float s0 = 0.f, q0 = 0.f, s1 = 0.f, q1 = 0.f;
#pragma unroll
    for (int r = 0; r < 8; ++r) {
        int mo = mtile * 16 + half * 8 + r;
        float t0v = acc0[r] + bv0;
        float t1v = acc1[r] + bv1;
        out[(size_t)mo * 256 + n0] = t0v;
        out[(size_t)mo * 256 + n1] = t1v;
        s0 += t0v; q0 += t0v * t0v;
        s1 += t1v; q1 += t1v * t1v;
    }
    // merge M=0..7 (lanes 0-15) with M=8..15 (lanes 16-31): same column per lane&15
    s0 += __shfl_down(s0, 16, 32); q0 += __shfl_down(q0, 16, 32);
    s1 += __shfl_down(s1, 16, 32); q1 += __shfl_down(q1, 16, 32);
    if (lane < 16) {
        atomicAdd(&csum[n0], s0); atomicAdd(&csq[n0], q0);
        atomicAdd(&csum[n1], s1); atomicAdd(&csq[n1], q1);
    }
}

// ---------------------------------------------------------------- GEMM2: z2[m,0:128] = relu(BN1(t[m,:])) @ W2 + b2
__global__ __launch_bounds__(256) void gemm2_kernel(
    const float* __restrict__ t,
    const float* __restrict__ scale, const float* __restrict__ shift,
    const v16bf* __restrict__ packB, const float* __restrict__ bias,
    float* __restrict__ out, float* __restrict__ csum, float* __restrict__ csq)
{
    __shared__ __bf16 As[16 * 256];  // 8 KB
    const int tid = threadIdx.x;
    const int lane = tid & 31;
    const int wave = tid >> 5;
    const int mtile = blockIdx.x;

    // stage A: relu(t*scale+shift) 16x256 -> bf16 LDS, 16 elements per thread
    {
        int e = tid * 16;
        int m = e >> 8, k = e & 255;
        const float* tp = t + (size_t)(mtile * 16 + m) * 256 + k;
        __bf16* dst = As + m * 256 + k;
#pragma unroll
        for (int q = 0; q < 4; ++q) {
            float4 tv = *(const float4*)(tp + 4 * q);
            float4 sc = *(const float4*)(scale + k + 4 * q);
            float4 sh = *(const float4*)(shift + k + 4 * q);
            dst[4 * q + 0] = (__bf16)fmaxf(tv.x * sc.x + sh.x, 0.f);
            dst[4 * q + 1] = (__bf16)fmaxf(tv.y * sc.y + sh.y, 0.f);
            dst[4 * q + 2] = (__bf16)fmaxf(tv.z * sc.z + sh.z, 0.f);
            dst[4 * q + 3] = (__bf16)fmaxf(tv.w * sc.w + sh.w, 0.f);
        }
    }
    __syncthreads();

    const int ntile = wave;          // 8 waves cover 128 cols
    const int mrow = lane & 15;
    const int half = lane >> 4;
    v8f acc = {};
#pragma unroll
    for (int ks = 0; ks < 8; ++ks) {
        v16bf a = load_a_frag(As, 256, mrow, half, ks * 32);
        v16bf b = packB[(ntile * 8 + ks) * 32 + lane];
        acc = __builtin_amdgcn_wmma_f32_16x16x32_bf16(false, a, false, b, (short)0, acc, false, false);
    }
    const int nn = ntile * 16 + (lane & 15);
    const float bv = bias[nn];
    float s = 0.f, q = 0.f;
#pragma unroll
    for (int r = 0; r < 8; ++r) {
        int mo = mtile * 16 + half * 8 + r;
        float zv = acc[r] + bv;
        out[(size_t)mo * DD + nn] = zv;
        s += zv; q += zv * zv;
    }
    s += __shfl_down(s, 16, 32); q += __shfl_down(q, 16, 32);
    if (lane < 16) {
        atomicAdd(&csum[nn], s); atomicAdd(&csq[nn], q);
    }
}

__global__ void bn_finalize_kernel(const float* __restrict__ sum, const float* __restrict__ sumsq,
                                   const float* __restrict__ g, const float* __restrict__ b,
                                   float* __restrict__ scale, float* __restrict__ shift)
{
    int c = threadIdx.x;
    float mean = sum[c] * (1.0f / NN);
    float var  = sumsq[c] * (1.0f / NN) - mean * mean;
    float inv  = rsqrtf(var + 1e-5f);
    float sc   = g[c] * inv;
    scale[c] = sc;
    shift[c] = b[c] - mean * sc;
}

// ---------------------------------------------------------------- h += maybe_relu(BN2(z2))
__global__ void update_h_kernel(float* __restrict__ h, const float* __restrict__ z2,
                                const float* __restrict__ scale, const float* __restrict__ shift,
                                int n4, int doRelu)
{
    int i = blockIdx.x * blockDim.x + threadIdx.x;
    if (i >= n4) return;
    int base = i * 4;
    int c = base & (DD - 1);
    float4 z = *(const float4*)(z2 + base);
    float4 hv = *(float4*)(h + base);
    float v0 = z.x * scale[c + 0] + shift[c + 0];
    float v1 = z.y * scale[c + 1] + shift[c + 1];
    float v2 = z.z * scale[c + 2] + shift[c + 2];
    float v3 = z.w * scale[c + 3] + shift[c + 3];
    if (doRelu) { v0 = fmaxf(v0, 0.f); v1 = fmaxf(v1, 0.f); v2 = fmaxf(v2, 0.f); v3 = fmaxf(v3, 0.f); }
    hv.x += v0; hv.y += v1; hv.z += v2; hv.w += v3;
    *(float4*)(h + base) = hv;
}

// ---------------------------------------------------------------- node attention + segment sums
__global__ __launch_bounds__(256) void node_att_kernel(
    const float* __restrict__ h, const float* __restrict__ Wt, const float* __restrict__ bp,
    const int* __restrict__ batch, float* __restrict__ out_key,
    float* __restrict__ key_num, float* __restrict__ env_num)
{
    int n = blockIdx.x * 8 + (threadIdx.x >> 5);
    if (n >= NN) return;
    int lane = threadIdx.x & 31;
    int c = lane * 4;
    const float4 hv = *(const float4*)(h + (size_t)n * DD + c);
    const float4 wv = *(const float4*)(Wt + c);
    float v = hv.x * wv.x + hv.y * wv.y + hv.z * wv.z + hv.w * wv.w;
#pragma unroll
    for (int off = 16; off > 0; off >>= 1) v += __shfl_down(v, off, 32);
    if (lane == 0) {
        float key = sigmoidf_(v + bp[0]);
        out_key[n] = key;
        int g = batch[n];
        atomicAdd(&key_num[g], key);
        atomicAdd(&env_num[g], 1.0f - key);
    }
}

// ---------------------------------------------------------------- edge attention + segment sums
__global__ __launch_bounds__(256) void edge_att_kernel(
    const float* __restrict__ h, const float* __restrict__ Wt, const float* __restrict__ bp,
    const int* __restrict__ edge_index, const int* __restrict__ batch,
    float* __restrict__ out_key, float* __restrict__ key_num, float* __restrict__ env_num)
{
    int e = blockIdx.x * 8 + (threadIdx.x >> 5);
    if (e >= EE) return;
    int lane = threadIdx.x & 31;
    int c = lane * 4;
    int row = edge_index[e];
    int col = edge_index[EE + e];
    const float4 hr = *(const float4*)(h + (size_t)row * DD + c);
    const float4 w0 = *(const float4*)(Wt + c);
    const float4 hc = *(const float4*)(h + (size_t)col * DD + c);
    const float4 w1 = *(const float4*)(Wt + DD + c);
    float v = hr.x * w0.x + hr.y * w0.y + hr.z * w0.z + hr.w * w0.w
            + hc.x * w1.x + hc.y * w1.y + hc.z * w1.z + hc.w * w1.w;
#pragma unroll
    for (int off = 16; off > 0; off >>= 1) v += __shfl_down(v, off, 32);
    if (lane == 0) {
        float key = sigmoidf_(v + bp[0]);
        out_key[e] = key;
        int g = batch[row];
        atomicAdd(&key_num[g], key);
        atomicAdd(&env_num[g], 1.0f - key);
    }
}

// ================================================================ launch
extern "C" void kernel_launch(void* const* d_in, const int* in_sizes, int n_in,
                              void* d_out, int out_size, void* d_ws, size_t ws_size,
                              hipStream_t stream) {
    const int*   x           = (const int*)d_in[0];
    const int*   edge_index  = (const int*)d_in[1];
    const int*   edge_attr   = (const int*)d_in[2];
    const int*   batch       = (const int*)d_in[3];
    const float* atom_tables = (const float*)d_in[4];
    const float* bond_tables = (const float*)d_in[5];
    const float* eps         = (const float*)d_in[6];
    const float* W1          = (const float*)d_in[7];
    const float* b1          = (const float*)d_in[8];
    const float* bn1_g       = (const float*)d_in[9];
    const float* bn1_b       = (const float*)d_in[10];
    const float* W2          = (const float*)d_in[11];
    const float* b2          = (const float*)d_in[12];
    const float* obn_g       = (const float*)d_in[13];
    const float* obn_b       = (const float*)d_in[14];
    const float* natt_W      = (const float*)d_in[15];
    const float* natt_b      = (const float*)d_in[16];
    const float* eatt_W      = (const float*)d_in[17];
    const float* eatt_b      = (const float*)d_in[18];

    float* out = (float*)d_out;
    float* node_key = out;
    float* edge_key = out + NN;
    float* nk_num   = out + NN + EE;
    float* nenv_num = nk_num + GG;
    float* ek_num   = nenv_num + GG;
    float* eenv_num = ek_num + GG;

    float* ws = (float*)d_ws;
    float* h      = ws;                            // N*D
    float* t      = ws + (size_t)NN * DD;          // N*2D
    float* agg    = ws + (size_t)NN * 3 * DD;      // N*D (reused as z2)
    float* sum1   = ws + (size_t)NN * 4 * DD;      // 256
    float* sq1    = sum1 + 256;
    float* sum2   = sq1 + 256;                     // 128
    float* sq2    = sum2 + 128;
    float* scale1 = sq2 + 128;                     // 256
    float* shift1 = scale1 + 256;
    float* scale2 = shift1 + 256;                  // 128
    float* shift2 = scale2 + 128;
    // packed bf16 B fragments (32B aligned: offsets are multiples of 8 floats)
    v16bf* packB1 = (v16bf*)(shift2 + 128);        // 2048 fragments = 64 KB
    v16bf* packB2 = (v16bf*)(shift2 + 128 + 16384);// 2048 fragments = 64 KB

    // init segment accumulators to 1e-8 (the reference's epsilon term)
    fill_kernel<<<(4 * GG + 255) / 256, 256, 0, stream>>>(nk_num, 4 * GG, 1e-8f);

    // AtomEncoder
    atom_encoder_kernel<<<NN / 8, 256, 0, stream>>>(x, atom_tables, h);

    for (int l = 0; l < 3; ++l) {
        const float* bond_l = bond_tables + (size_t)l * 3 * BV * DD;

        // pack this layer's weights into WMMA fragment order (bf16)
        pack_b_kernel<<<8, 256, 0, stream>>>(W1 + (size_t)l * DD * 256, 256, 4, packB1);
        pack_b_kernel<<<8, 256, 0, stream>>>(W2 + (size_t)l * 256 * DD, DD, 8, packB2);

        fill_kernel<<<((NN * DD) + 255) / 256, 256, 0, stream>>>(agg, NN * DD, 0.f);
        message_kernel<<<EE / 8, 256, 0, stream>>>(h, edge_index, edge_attr, bond_l, agg);

        // GEMM1 (fused (1+eps)h + agg; BN1 stats fused into epilogue)
        fill_kernel<<<(512 + 255) / 256, 256, 0, stream>>>(sum1, 512, 0.f);
        gemm1_kernel<<<NN / 16, 256, 0, stream>>>(h, agg, eps + l, packB1,
                                                  b1 + (size_t)l * 256, t, sum1, sq1);
        bn_finalize_kernel<<<1, 256, 0, stream>>>(sum1, sq1, bn1_g + (size_t)l * 256,
                                                  bn1_b + (size_t)l * 256, scale1, shift1);

        // GEMM2 (fused BN1 affine + relu on A; BN2 stats fused into epilogue)
        fill_kernel<<<(256 + 255) / 256, 256, 0, stream>>>(sum2, 256, 0.f);
        gemm2_kernel<<<NN / 16, 256, 0, stream>>>(t, scale1, shift1, packB2,
                                                  b2 + (size_t)l * DD, agg, sum2, sq2);
        bn_finalize_kernel<<<1, 128, 0, stream>>>(sum2, sq2, obn_g + (size_t)l * DD,
                                                  obn_b + (size_t)l * DD, scale2, shift2);

        // residual update (relu on layers 0,1 only)
        int n4 = NN * DD / 4;
        update_h_kernel<<<(n4 + 255) / 256, 256, 0, stream>>>(h, agg, scale2, shift2, n4,
                                                              (l < 2) ? 1 : 0);
    }

    node_att_kernel<<<NN / 8, 256, 0, stream>>>(h, natt_W, natt_b, batch,
                                                node_key, nk_num, nenv_num);
    edge_att_kernel<<<EE / 8, 256, 0, stream>>>(h, eatt_W, eatt_b, edge_index, batch,
                                                edge_key, ek_num, eenv_num);
    (void)in_sizes; (void)n_in; (void)out_size; (void)ws_size;
}
